// Attention_1391569404051
// MI455X (gfx1250) — matrix operand order repaired and checked by reading, not compiled, not run on hardware
//
#include <hip/hip_runtime.h>
#include <stdint.h>

typedef __attribute__((ext_vector_type(16))) _Float16 v16h;
typedef __attribute__((ext_vector_type(8)))  float    v8f;
typedef unsigned int u32x4v __attribute__((ext_vector_type(4)));
typedef int          i32x8v __attribute__((ext_vector_type(8)));
typedef int          i32x4v __attribute__((ext_vector_type(4)));

#define DIM       512
#define HIDDEN    512
#define HEADS     8
#define DHEAD     64
#define NTOK      16384      // B*N = 8*2048
#define SEQ       2048
#define LDK_X     32         // LDS k-stride for A tiles (halves)
#define LDK_W     40         // LDS k-stride for B tiles (halves, padded)
#define LDK_KV    72         // LDS stride for K/V tiles (halves, padded)

// ds_swizzle_b32 XOR exchange (group-of-32 mode: offset[14:10]=xor, [4:0]=and)
// One DS op, no VALU address math (vs ds_bpermute lowering of __shfl_xor).
#define XOR_SWZ(val, mask)                                                    \
  __uint_as_float((uint32_t)__builtin_amdgcn_ds_swizzle(                      \
      (int)__float_as_uint(val), (((mask) << 10) | 31)))

__device__ inline float half_sum16(float v) {   // reduce+bcast over 16 lanes
  v += XOR_SWZ(v, 1); v += XOR_SWZ(v, 2); v += XOR_SWZ(v, 4); v += XOR_SWZ(v, 8);
  return v;
}
__device__ inline float half_max16(float v) {
  v = fmaxf(v, XOR_SWZ(v, 1)); v = fmaxf(v, XOR_SWZ(v, 2));
  v = fmaxf(v, XOR_SWZ(v, 4)); v = fmaxf(v, XOR_SWZ(v, 8));
  return v;
}
__device__ inline float wave_sum32(float v) {
  v = half_sum16(v); v += XOR_SWZ(v, 16);
  return v;
}

// ---------------------------------------------------------------------------
// WMMA fragment loaders (CDNA5 16-bit layouts, CDNA5 ISA §7.12.2)
// A 16x32 (MxK): lane m=l%16; VGPR j, half p: k = (j/4)*16 + (l/16)*8 + (j%4)*2 + p
// B 32x16 (KxN): lane n=l%16; VGPR j, half p: k = (j/4)*16 + (l/16)*8 + (j%4)*2 + p
// C/D 16x16 f32: lane n=l%16; VGPR r: m = (l/16)*8 + r
// LDS tiles stored so that k runs contiguously (A: [m][k], B: [n][k]).
// ---------------------------------------------------------------------------
__device__ inline v16h load_A(const _Float16* lds, int lane, int ld) {
  const int m  = lane & 15;
  const int hi = lane >> 4;
  union { v16h v; uint32_t u[8]; } a;
#pragma unroll
  for (int j = 0; j < 8; ++j) {
    int k = ((j >> 2) << 4) + (hi << 3) + ((j & 3) << 1);
    a.u[j] = *(const uint32_t*)(lds + m * ld + k);
  }
  return a.v;
}

__device__ inline v16h load_B(const _Float16* lds, int lane, int ld) {
  const int n  = lane & 15;
  const int hi = lane >> 4;
  union { v16h v; uint32_t u[8]; } b;
#pragma unroll
  for (int j = 0; j < 8; ++j) {
    int k = ((j >> 2) << 4) + (hi << 3) + ((j & 3) << 1);
    b.u[j] = *(const uint32_t*)(lds + n * ld + k);
  }
  return b.v;
}

__device__ inline v8f zero8() { v8f z = {0.f,0.f,0.f,0.f,0.f,0.f,0.f,0.f}; return z; }

// ---------------------------------------------------------------------------
// Kernel 1: RMSNorm  xn = x / max(||x||,eps) * g * sqrt(512)  ->  f16
// ---------------------------------------------------------------------------
__global__ __launch_bounds__(256) void rmsnorm_kernel(
    const float* __restrict__ x, const float* __restrict__ g,
    _Float16* __restrict__ XN) {
  __shared__ float red[8];
  const int row = blockIdx.x;
  const float2 v = ((const float2*)(x + (size_t)row * DIM))[threadIdx.x];
  float ss = wave_sum32(v.x * v.x + v.y * v.y);
  const int lane = threadIdx.x & 31, wv = threadIdx.x >> 5;
  if (lane == 0) red[wv] = ss;
  __syncthreads();
  float tot = red[0] + red[1] + red[2] + red[3] +
              red[4] + red[5] + red[6] + red[7];
  const float scale = 22.6274169979695f / fmaxf(sqrtf(tot), 1e-12f); // sqrt(512)
  const int d = threadIdx.x * 2;
  XN[(size_t)row * DIM + d]     = (_Float16)(v.x * scale * g[d]);
  XN[(size_t)row * DIM + d + 1] = (_Float16)(v.y * scale * g[d + 1]);
}

// ---------------------------------------------------------------------------
// Kernel 2: QKV GEMM [16384,512]x[512,1536] with fused per-head l2norm of q,k.
// grid (128, 24): block = 128 tokens x 64 cols (one head-slice of q|k|v).
// ---------------------------------------------------------------------------
__global__ __launch_bounds__(256) void qkv_gemm_kernel(
    const _Float16* __restrict__ XN, const float* __restrict__ Wqkv,
    const float* __restrict__ q_scale, const float* __restrict__ k_scale,
    _Float16* __restrict__ Q, _Float16* __restrict__ K, _Float16* __restrict__ V) {
  __shared__ __align__(16) _Float16 ldsX[128 * LDK_X];
  __shared__ __align__(16) _Float16 ldsW[64 * LDK_W];
  const int tid = threadIdx.x, lane = tid & 31, w = tid >> 5;
  const int tb = blockIdx.x * 128;
  const int cb = blockIdx.y;            // 0..7 q, 8..15 k, 16..23 v
  const int col0 = cb * 64;

  v8f acc[4];
#pragma unroll
  for (int t = 0; t < 4; ++t) acc[t] = zero8();

  for (int kb = 0; kb < DIM; kb += 32) {
    __syncthreads();
    {
      const uint32_t* src = (const uint32_t*)XN;
#pragma unroll
      for (int i = 0; i < 8; ++i) {
        int u = tid + i * 256;
        int row = u >> 4, cu = u & 15;
        ((uint32_t*)ldsX)[row * (LDK_X / 2) + cu] =
            src[(size_t)(tb + row) * (DIM / 2) + (kb >> 1) + cu];
      }
    }
    {
#pragma unroll
      for (int i = 0; i < 8; ++i) {
        int e = tid + i * 256;
        int n = e & 63, kk = e >> 6;
        ldsW[n * LDK_W + kk] = (_Float16)Wqkv[(size_t)(kb + kk) * 1536 + col0 + n];
      }
    }
    __syncthreads();
    v16h a = load_A(ldsX + (w * 16) * LDK_X, lane, LDK_X);
#pragma unroll
    for (int t = 0; t < 4; ++t) {
      v16h b = load_B(ldsW + (t * 16) * LDK_W, lane, LDK_W);
      acc[t] = __builtin_amdgcn_wmma_f32_16x16x32_f16(
          false, a, false, b, (short)0, acc[t], false, false);
    }
  }

  const int lane15 = lane & 15, hi = lane >> 4;
  // fused l2norm over the 64-dim head for q/k (SCALE=8 folded into q)
  if (cb < 16) {
    const float extra = (cb < 8) ? 8.0f : 1.0f;
#pragma unroll
    for (int r = 0; r < 8; ++r) {
      float ss = 0.f;
#pragma unroll
      for (int t = 0; t < 4; ++t) { float vv = acc[t][r]; ss += vv * vv; }
      ss = half_sum16(ss);
      float inv = extra / fmaxf(sqrtf(ss), 1e-12f);
#pragma unroll
      for (int t = 0; t < 4; ++t) acc[t][r] *= inv;
    }
  }
  _Float16* dst; int head;
  if (cb < 8)       { dst = Q; head = cb; }
  else if (cb < 16) { dst = K; head = cb - 8; }
  else              { dst = V; head = cb - 16; }
#pragma unroll
  for (int t = 0; t < 4; ++t) {
    int d = t * 16 + lane15;
    float sc = (cb < 8) ? q_scale[d] : (cb < 16 ? k_scale[d] : 1.0f);
#pragma unroll
    for (int r = 0; r < 8; ++r) {
      int gt = tb + w * 16 + hi * 8 + r;
      int b = gt >> 11, nt = gt & 2047;
      dst[((size_t)(b * HEADS + head) * SEQ + nt) * DHEAD + d] =
          (_Float16)(acc[t][r] * sc);
    }
  }
}

// ---------------------------------------------------------------------------
// Kernel 3: flash attention per (b,h).  grid (16, 64); wave owns 16 queries.
// K tile: TDM tensor_load_to_lds with LDS padding (64x64 f16 tile, pad 4
//   dwords every 32 dwords -> padded stride LDK_KV=72 halves).
// V tile: manual transposed stage [d][key].  P: C->A layout via per-wave LDS.
// ---------------------------------------------------------------------------
__global__ __launch_bounds__(256) void attn_kernel(
    const _Float16* __restrict__ Q, const _Float16* __restrict__ Kb,
    const _Float16* __restrict__ Vb, _Float16* __restrict__ O) {
  __shared__ __align__(16) _Float16 ldsK[64 * LDK_KV];
  __shared__ __align__(16) _Float16 ldsV[64 * LDK_KV];
  __shared__ __align__(16) _Float16 ldsP[8 * 16 * 64];
  const int tid = threadIdx.x, lane = tid & 31, w = tid >> 5;
  const int bh = blockIdx.y;
  const int q0 = blockIdx.x * 128 + w * 16;
  const size_t base = (size_t)bh * SEQ * DHEAD;
  const _Float16* Qh = Q + base;
  const _Float16* Kh = Kb + base;
  const _Float16* Vh = Vb + base;
  const int n0 = lane & 15, hi = lane >> 4;

  // Q A-fragments (two 16x32 chunks over d), loaded straight from global
  v16h aq[2];
#pragma unroll
  for (int c = 0; c < 2; ++c) {
    union { v16h v; uint32_t u[8]; } tmp;
#pragma unroll
    for (int j = 0; j < 8; ++j) {
      int k = c * 32 + ((j >> 2) << 4) + (hi << 3) + ((j & 3) << 1);
      tmp.u[j] = *(const uint32_t*)(Qh + (size_t)(q0 + n0) * DHEAD + k);
    }
    aq[c] = tmp.v;
  }

  float Mrow[8], Lrow[8];
  v8f oacc[4];
#pragma unroll
  for (int r = 0; r < 8; ++r) { Mrow[r] = -1e30f; Lrow[r] = 0.f; }
#pragma unroll
  for (int t = 0; t < 4; ++t) oacc[t] = zero8();
  _Float16* pW = ldsP + w * (16 * 64);

  // TDM descriptor group1 is loop-invariant (CDNA5 ISA):
  //   data_size=1 (2B), pad_enable=1, pad_interval=4 (32 dwords = one 64-half
  //   row), pad_amount=3 (4 dwords = 8 halves) -> LDS row stride 72 halves.
  //   tensor_dim0=64, tensor_dim1=2048, tile_dim0=64, tile_dim1=64,
  //   tensor_dim0_stride=64.
  i32x8v tdm_g1;
  tdm_g1[0] = (1 << 16) | (1 << 20) | (4 << 22) | (3 << 25);
  tdm_g1[1] = (int)(64u << 16);          // tensor_dim0[15:0] @ bits 63:48
  tdm_g1[2] = (int)(2048u << 16);        // dim0[31:16]=0 | tensor_dim1[15:0]
  tdm_g1[3] = (int)(64u << 16);          // dim1[31:16]=0 | tile_dim0=64
  tdm_g1[4] = 64;                        // tile_dim1=64, tile_dim2=0
  tdm_g1[5] = 64;                        // tensor_dim0_stride[31:0]
  tdm_g1[6] = 0;
  tdm_g1[7] = 0;
  const i32x4v tdm_zero4 = {0, 0, 0, 0};
  const i32x8v tdm_zero8 = {0, 0, 0, 0, 0, 0, 0, 0};
  const uint32_t ldsK_off = (uint32_t)(uintptr_t)(&ldsK[0]);

  for (int j0 = 0; j0 < SEQ; j0 += 64) {
    __syncthreads();
    // ---- K tile via Tensor Data Mover (wave 0 issues; TDM ignores EXEC) ----
    if (w == 0) {
      uint64_t ga = (uint64_t)(uintptr_t)(Kh + (size_t)j0 * DHEAD);
      u32x4v g0;
      g0[0] = 1u;                                   // count=1 (valid D#)
      g0[1] = ldsK_off;                             // lds_addr
      g0[2] = (uint32_t)ga;                         // global_addr[31:0]
      g0[3] = (uint32_t)((ga >> 32) & 0x1FFFFFFu)   // global_addr[56:32]
              | (2u << 30);                         // type=2 ("image")
      __builtin_amdgcn_tensor_load_to_lds(g0, tdm_g1, tdm_zero4, tdm_zero4,
                                          tdm_zero8, 0);
    }
    // ---- V tile transposed [d][key] (all threads) ----
    {
#pragma unroll
      for (int i = 0; i < 16; ++i) {
        int e = tid + i * 256;
        int d = e & 63, key = e >> 6;
        ldsV[d * LDK_KV + key] = Vh[(size_t)(j0 + key) * DHEAD + d];
      }
    }
    if (w == 0) __builtin_amdgcn_s_wait_tensorcnt(0);
    __syncthreads();

    // S = Q K^T  (16 queries x 64 keys; x8 scale folded into q)
    v8f s[4];
#pragma unroll
    for (int t = 0; t < 4; ++t) {
      v8f sv = zero8();
#pragma unroll
      for (int c = 0; c < 2; ++c) {
        v16h b = load_B(ldsK + (t * 16) * LDK_KV + c * 32, lane, LDK_KV);
        sv = __builtin_amdgcn_wmma_f32_16x16x32_f16(
            false, aq[c], false, b, (short)0, sv, false, false);
      }
      s[t] = sv;
    }

    // online softmax (per-row stats via ds_swizzle xor exchanges)
#pragma unroll
    for (int r = 0; r < 8; ++r) {
      float mx = fmaxf(fmaxf(s[0][r], s[1][r]), fmaxf(s[2][r], s[3][r]));
      mx = half_max16(mx);
      float Mn = fmaxf(Mrow[r], mx);
      float alpha = __expf(Mrow[r] - Mn);
      Mrow[r] = Mn;
      float ls = 0.f;
#pragma unroll
      for (int t = 0; t < 4; ++t) {
        float p = __expf(s[t][r] - Mn);
        s[t][r] = p;
        ls += p;
      }
      ls = half_sum16(ls);
      Lrow[r] = Lrow[r] * alpha + ls;
#pragma unroll
      for (int t = 0; t < 4; ++t) oacc[t][r] *= alpha;
    }

    // P: C-layout -> per-wave LDS [m][key] -> A-layout
#pragma unroll
    for (int t = 0; t < 4; ++t)
#pragma unroll
      for (int r = 0; r < 8; ++r)
        pW[(hi * 8 + r) * 64 + t * 16 + n0] = (_Float16)s[t][r];

    v16h ap0 = load_A(pW, lane, 64);
    v16h ap1 = load_A(pW + 32, lane, 64);
#pragma unroll
    for (int t = 0; t < 4; ++t) {
      v16h b0 = load_B(ldsV + (t * 16) * LDK_KV, lane, LDK_KV);
      oacc[t] = __builtin_amdgcn_wmma_f32_16x16x32_f16(
          false, ap0, false, b0, (short)0, oacc[t], false, false);
      v16h b1 = load_B(ldsV + (t * 16) * LDK_KV + 32, lane, LDK_KV);
      oacc[t] = __builtin_amdgcn_wmma_f32_16x16x32_f16(
          false, ap1, false, b1, (short)0, oacc[t], false, false);
    }
  }

  // finalize: O[b, tok, h*64 + d] = oacc / L
  const int b = bh >> 3, h = bh & 7;
#pragma unroll
  for (int r = 0; r < 8; ++r) {
    float invL = 1.0f / Lrow[r];
    int tok = q0 + hi * 8 + r;
#pragma unroll
    for (int t = 0; t < 4; ++t) {
      int d = t * 16 + n0;
      O[(size_t)(b * SEQ + tok) * HIDDEN + h * DHEAD + d] =
          (_Float16)(oacc[t][r] * invL);
    }
  }
}

// ---------------------------------------------------------------------------
// Kernel 4: out = O[16384,512] @ w_out[512,512]  -> f32
// ---------------------------------------------------------------------------
__global__ __launch_bounds__(256) void out_gemm_kernel(
    const _Float16* __restrict__ O, const float* __restrict__ Wout,
    float* __restrict__ out) {
  __shared__ __align__(16) _Float16 ldsX[128 * LDK_X];
  __shared__ __align__(16) _Float16 ldsW[64 * LDK_W];
  const int tid = threadIdx.x, lane = tid & 31, w = tid >> 5;
  const int tb = blockIdx.x * 128;
  const int col0 = blockIdx.y * 64;

  v8f acc[4];
#pragma unroll
  for (int t = 0; t < 4; ++t) acc[t] = zero8();

  for (int kb = 0; kb < HIDDEN; kb += 32) {
    __syncthreads();
    {
      const uint32_t* src = (const uint32_t*)O;
#pragma unroll
      for (int i = 0; i < 8; ++i) {
        int u = tid + i * 256;
        int row = u >> 4, cu = u & 15;
        ((uint32_t*)ldsX)[row * (LDK_X / 2) + cu] =
            src[(size_t)(tb + row) * (HIDDEN / 2) + (kb >> 1) + cu];
      }
    }
    {
#pragma unroll
      for (int i = 0; i < 8; ++i) {
        int e = tid + i * 256;
        int n = e & 63, kk = e >> 6;
        ldsW[n * LDK_W + kk] = (_Float16)Wout[(size_t)(kb + kk) * DIM + col0 + n];
      }
    }
    __syncthreads();
    v16h a = load_A(ldsX + (w * 16) * LDK_X, lane, LDK_X);
#pragma unroll
    for (int t = 0; t < 4; ++t) {
      v16h b = load_B(ldsW + (t * 16) * LDK_W, lane, LDK_W);
      acc[t] = __builtin_amdgcn_wmma_f32_16x16x32_f16(
          false, a, false, b, (short)0, acc[t], false, false);
    }
  }

  const int lane15 = lane & 15, hi = lane >> 4;
#pragma unroll
  for (int t = 0; t < 4; ++t)
#pragma unroll
    for (int r = 0; r < 8; ++r)
      out[(size_t)(tb + w * 16 + hi * 8 + r) * DIM + col0 + t * 16 + lane15] =
          acc[t][r];
}

// ---------------------------------------------------------------------------
extern "C" void kernel_launch(void* const* d_in, const int* in_sizes, int n_in,
                              void* d_out, int out_size, void* d_ws, size_t ws_size,
                              hipStream_t stream) {
  const float* x       = (const float*)d_in[0];
  const float* g       = (const float*)d_in[1];
  const float* w_qkv   = (const float*)d_in[2];
  const float* q_scale = (const float*)d_in[3];
  const float* k_scale = (const float*)d_in[4];
  const float* w_out   = (const float*)d_in[5];
  float* out = (float*)d_out;

  const size_t SLAB = (size_t)NTOK * 512 * sizeof(_Float16); // 16 MiB
  char* ws = (char*)d_ws;
  _Float16* XN = (_Float16*)(ws + 0 * SLAB);
  _Float16* Qb = (_Float16*)(ws + 1 * SLAB);
  _Float16* Kb = (_Float16*)(ws + 2 * SLAB);
  _Float16* Vb = (_Float16*)(ws + 3 * SLAB);
  _Float16* Ob = (_Float16*)(ws + 4 * SLAB);

  rmsnorm_kernel<<<NTOK, 256, 0, stream>>>(x, g, XN);
  qkv_gemm_kernel<<<dim3(NTOK / 128, 24), 256, 0, stream>>>(
      XN, w_qkv, q_scale, k_scale, Qb, Kb, Vb);
  attn_kernel<<<dim3(SEQ / 128, 64), 256, 0, stream>>>(Qb, Kb, Vb, Ob);
  out_gemm_kernel<<<dim3(NTOK / 128, DIM / 64), 256, 0, stream>>>(Ob, w_out, out);
}
